// BRITSNet_48326972015003
// MI455X (gfx1250) — matrix operand order repaired and checked by reading, not compiled, not run on hardware
//
#include <hip/hip_runtime.h>

// ---------------------------------------------------------------------------
// BRITS fused bidirectional RITS scan for gfx1250 (MI455X).
// One workgroup = (direction, 16-row batch tile); weights resident in LDS as
// f16; all per-step GEMMs via V_WMMA_F32_16X16X32_F16; LSTM/decay elementwise
// in fp32. Grid = 2 dirs * 16 batch tiles = 32 blocks * 256 threads (8 waves).
// ---------------------------------------------------------------------------

typedef _Float16 h16;
typedef __attribute__((ext_vector_type(16))) _Float16 v16h;
typedef __attribute__((ext_vector_type(8)))  _Float16 v8h;
typedef __attribute__((ext_vector_type(8)))  float    v8f;

namespace brits {

constexpr int B = 256, T = 1024, D = 64, H = 64, G = 256; // G = 4*H
constexpr int NT = 256;

// ---- LDS layout: f16 region (element offsets) ----
constexpr int OFF_WIH  = 0;                     // [256][128]
constexpr int OFF_WHH  = OFF_WIH  + G * 2 * D;  // [256][64]
constexpr int OFF_WDH  = OFF_WHH  + G * H;      // [64][64]
constexpr int OFF_WHR  = OFF_WDH  + H * D;      // [64][64]
constexpr int OFF_WFR  = OFF_WHR  + D * H;      // [64][64] (zero diag)
constexpr int OFF_WWC  = OFF_WFR  + D * D;      // [64][128]
constexpr int OFF_H16  = OFF_WWC  + D * 2 * D;  // [16][64] h (pre-decay)
constexpr int OFF_HDEC = OFF_H16  + 16 * H;     // [16][64] h*gamma_h
constexpr int OFF_DEL  = OFF_HDEC + 16 * H;     // [16][64] delta
constexpr int OFF_XC   = OFF_DEL  + 16 * D;     // [16][64] x_c
constexpr int OFF_CAT1 = OFF_XC   + 16 * D;     // [16][128] [gamma_x | m]
constexpr int OFF_CAT2 = OFF_CAT1 + 16 * 2 * D; // [16][128] [c_c | m]
constexpr int HALF_N   = OFF_CAT2 + 16 * 2 * D;

// ---- LDS layout: f32 region (element offsets) ----
constexpr int F_H32   = 0;                 // [16][64]
constexpr int F_GAM   = F_H32   + 16 * H;  // [16][64] gamma_h
constexpr int F_XH    = F_GAM   + 16 * H;  // [16][64] x_h
constexpr int F_ZH    = F_XH    + 16 * D;  // [16][64] z_h
constexpr int F_AL    = F_ZH    + 16 * D;  // [16][64] alpha
constexpr int F_GATES = F_AL    + 16 * D;  // [16][256]
constexpr int F_BIHH  = F_GATES + 16 * G;  // [256] bih+bhh
constexpr int F_BDH   = F_BIHH  + G;       // [64]
constexpr int F_BHR   = F_BDH   + H;       // [64]
constexpr int F_BFR   = F_BHR   + D;       // [64]
constexpr int F_BWC   = F_BFR   + D;       // [64]
constexpr int FLOAT_N = F_BWC   + D;

constexpr unsigned SMEM_BYTES = (unsigned)HALF_N * 2u + (unsigned)FLOAT_N * 4u;

__device__ __forceinline__ float sigm(float v) {
  return 1.0f / (1.0f + __expf(-v));
}

// A fragment: 16xK f16 activations, row-major [16][LDK] in LDS.
// A layout (matches B): lanes 0-15 -> M=lane, K=0..15;
// lanes 16-31 -> M=lane-16, K=16..31 (per 32-K slab).
__device__ __forceinline__ v16h frag_a(const h16* act, int ldk, int kbase, int lane) {
  const int row = lane & 15;
  const int off = (lane & 16) ? 16 : 0;
  const h16* p = act + row * ldk + kbase + off;
  return *(const v16h*)(p);
}

// B fragment: weight W stored row-major [N][LDK] f16 (y = x @ W.T).
// B layout (matches A): lane -> N=lane&15; lanes 0-15 hold K=kbase..kbase+15,
// lanes 16-31 hold K=kbase+16..kbase+31 -> one contiguous 32B LDS read.
__device__ __forceinline__ v16h frag_b(const h16* w, int ldk, int n0, int kbase, int lane) {
  const int n = n0 + (lane & 15);
  const int koff = kbase + ((lane & 16) ? 16 : 0);
  return *(const v16h*)(w + n * ldk + koff);
}

template <int K, int LDK>
__device__ __forceinline__ v8f wmma_gemm(v8f c, const h16* act, const h16* w,
                                         int n0, int lane) {
#pragma unroll
  for (int k = 0; k < K; k += 32) {
    v16h a = frag_a(act, LDK, k, lane);
    v16h b = frag_b(w, LDK, n0, k, lane);
    c = __builtin_amdgcn_wmma_f32_16x16x32_f16(false, a, false, b,
                                               (short)0, c, false, false);
  }
  return c;
}

__device__ __forceinline__ v8f tile_bias(const float* bias, int n0, int lane) {
  const float bv = bias[n0 + (lane & 15)];
  v8f c;
#pragma unroll
  for (int r = 0; r < 8; ++r) c[r] = bv;
  return c;
}

// MODE: 0 = plain, 1 = exp(-relu(v)) (temporal decay), 2 = sigmoid
template <int MODE>
__device__ __forceinline__ void tile_store(float* o, int ldn, int n0, v8f c, int lane) {
  const int col = n0 + (lane & 15);
  const int rb = (lane & 16) ? 8 : 0;
#pragma unroll
  for (int r = 0; r < 8; ++r) {
    float v = c[r];
    if (MODE == 1) v = __expf(-fmaxf(v, 0.0f));
    if (MODE == 2) v = 1.0f / (1.0f + __expf(-v));
    o[(rb + r) * ldn + col] = v;
  }
}

} // namespace brits

using namespace brits;

__launch_bounds__(brits::NT, 1)
__global__ void brits_fused(
    const float* __restrict__ xg,  const float* __restrict__ mg,
    const float* __restrict__ Wih, const float* __restrict__ Whh,
    const float* __restrict__ bih, const float* __restrict__ bhh,
    const float* __restrict__ Wdh, const float* __restrict__ bdh,
    const float* __restrict__ wdx, const float* __restrict__ bdx,
    const float* __restrict__ Whr, const float* __restrict__ bhr,
    const float* __restrict__ Wfr, const float* __restrict__ bfr,
    const float* __restrict__ Wwc, const float* __restrict__ bwc,
    float* __restrict__ out, float* __restrict__ pred)
{
  extern __shared__ __align__(16) char smem_raw[];
  h16*   sh = (h16*)smem_raw;
  float* sf = (float*)(smem_raw + (size_t)HALF_N * 2);

  const int tid  = threadIdx.x;
  const int lane = tid & 31;
  const int wv   = tid >> 5;       // wave id 0..7 (wave32)
  const int dir  = blockIdx.x >> 4;
  const int bt   = blockIdx.x & 15;

  // ---- one-time: weights -> LDS f16, biases -> LDS f32 ----
  {
    const float* g;
    g = Wih + (size_t)dir * G * 2 * D;
    for (int i = tid; i < G * 2 * D; i += NT) sh[OFF_WIH + i] = (h16)g[i];
    g = Whh + (size_t)dir * G * H;
    for (int i = tid; i < G * H; i += NT) sh[OFF_WHH + i] = (h16)g[i];
    g = Wdh + (size_t)dir * H * D;
    for (int i = tid; i < H * D; i += NT) sh[OFF_WDH + i] = (h16)g[i];
    g = Whr + (size_t)dir * D * H;
    for (int i = tid; i < D * H; i += NT) sh[OFF_WHR + i] = (h16)g[i];
    g = Wfr + (size_t)dir * D * D;
    for (int i = tid; i < D * D; i += NT) {
      const int r = i >> 6, c = i & 63;
      sh[OFF_WFR + i] = (h16)((r == c) ? 0.0f : g[i]);  // zero-diag feat_reg
    }
    g = Wwc + (size_t)dir * D * 2 * D;
    for (int i = tid; i < D * 2 * D; i += NT) sh[OFF_WWC + i] = (h16)g[i];
    for (int i = tid; i < G; i += NT)
      sf[F_BIHH + i] = bih[dir * G + i] + bhh[dir * G + i];
    if (tid < H) sf[F_BDH + tid] = bdh[dir * H + tid];
    if (tid < D) sf[F_BHR + tid] = bhr[dir * D + tid];
    if (tid < D) sf[F_BFR + tid] = bfr[dir * D + tid];
    if (tid < D) sf[F_BWC + tid] = bwc[dir * D + tid];
  }

  // per-thread elementwise slice: 16 rows x 64 feats / 256 threads = 4 feats
  const int bl = tid >> 4;          // local batch row 0..15
  const int d0 = (tid & 15) * 4;    // feature base
  const int b  = bt * 16 + bl;      // global batch row
  const int ei = bl * 64 + d0;      // index into [16][64] LDS tiles

  float c4[4]  = {0.f, 0.f, 0.f, 0.f};  // LSTM cell state (registers)
  float dl4[4] = {0.f, 0.f, 0.f, 0.f};  // temporal-decay delta (registers)
  float wdx4[4], bdx4[4];
#pragma unroll
  for (int j = 0; j < 4; ++j) {
    wdx4[j] = wdx[dir * D + d0 + j];
    bdx4[j] = bdx[dir * D + d0 + j];
  }
#pragma unroll
  for (int j = 0; j < 4; ++j) {
    sf[F_H32 + ei + j] = 0.0f;
    sh[OFF_H16 + ei + j] = (h16)0.0f;
  }
  __syncthreads();

  for (int t = 0; t < T; ++t) {
    const int tg = dir ? (T - 1 - t) : t;   // time index in original order
    const size_t tbase = ((size_t)b * T + tg) * D + d0;
    const float4 xv = *(const float4*)(xg + tbase);
    const float4 mv = *(const float4*)(mg + tbase);
    const float x4[4] = {xv.x, xv.y, xv.z, xv.w};
    const float m4[4] = {mv.x, mv.y, mv.z, mv.w};
    if (t + 1 < T) {  // prefetch next step's rows (global_prefetch_b8)
      const int tn = dir ? (T - 2 - t) : (t + 1);
      const size_t nb = ((size_t)b * T + tn) * D + d0;
      __builtin_prefetch(xg + nb, 0, 0);
      __builtin_prefetch(mg + nb, 0, 0);
    }
#pragma unroll
    for (int j = 0; j < 4; ++j) {
      sh[OFF_DEL + ei + j] = (h16)dl4[j];
      sh[OFF_CAT1 + bl * 128 + 64 + d0 + j] = (h16)m4[j];
      sh[OFF_CAT2 + bl * 128 + 64 + d0 + j] = (h16)m4[j];
    }
    __syncthreads();

    // stage A: gamma_h = exp(-relu(delta@Wdh.T+bdh)) | x_h = h@Whr.T+bhr
    if (wv < 4) {
      const int n0 = wv * 16;
      v8f c = tile_bias(sf + F_BDH, n0, lane);
      c = wmma_gemm<64, 64>(c, sh + OFF_DEL, sh + OFF_WDH, n0, lane);
      tile_store<1>(sf + F_GAM, 64, n0, c, lane);
    } else {
      const int n0 = (wv - 4) * 16;
      v8f c = tile_bias(sf + F_BHR, n0, lane);
      c = wmma_gemm<64, 64>(c, sh + OFF_H16, sh + OFF_WHR, n0, lane);
      tile_store<0>(sf + F_XH, 64, n0, c, lane);
    }
    __syncthreads();

    // stage B: x_c, h*gamma_h, gamma_x, delta update
#pragma unroll
    for (int j = 0; j < 4; ++j) {
      const float xh = sf[F_XH + ei + j];
      const float xc = m4[j] * x4[j] + (1.0f - m4[j]) * xh;
      sh[OFF_XC + ei + j] = (h16)xc;
      const float hd = sf[F_H32 + ei + j] * sf[F_GAM + ei + j];
      sh[OFF_HDEC + ei + j] = (h16)hd;
      const float gx = __expf(-fmaxf(dl4[j] * wdx4[j] + bdx4[j], 0.0f));
      sh[OFF_CAT1 + bl * 128 + d0 + j] = (h16)gx;
      dl4[j] = 1.0f + (1.0f - m4[j]) * dl4[j];  // delta for next step
    }
    __syncthreads();

    // stage C: z_h = x_c@WfrM.T+bfr | alpha = sigmoid([gx,m]@Wwc.T+bwc)
    if (wv < 4) {
      const int n0 = wv * 16;
      v8f c = tile_bias(sf + F_BFR, n0, lane);
      c = wmma_gemm<64, 64>(c, sh + OFF_XC, sh + OFF_WFR, n0, lane);
      tile_store<0>(sf + F_ZH, 64, n0, c, lane);
    } else {
      const int n0 = (wv - 4) * 16;
      v8f c = tile_bias(sf + F_BWC, n0, lane);
      c = wmma_gemm<128, 128>(c, sh + OFF_CAT1, sh + OFF_WWC, n0, lane);
      tile_store<2>(sf + F_AL, 64, n0, c, lane);
    }
    __syncthreads();

    // stage D: c_h, c_c; emit per-step outputs
    float chv[4], zhv[4], xhv[4], ccv[4];
#pragma unroll
    for (int j = 0; j < 4; ++j) {
      const float xh = sf[F_XH + ei + j];
      const float zh = sf[F_ZH + ei + j];
      const float al = sf[F_AL + ei + j];
      const float ch = al * zh + (1.0f - al) * xh;
      const float cc = m4[j] * x4[j] + (1.0f - m4[j]) * ch;
      sh[OFF_CAT2 + bl * 128 + d0 + j] = (h16)cc;
      chv[j] = ch; zhv[j] = zh; xhv[j] = xh; ccv[j] = cc;
    }
    {
      constexpr size_t PS = (size_t)B * T * D;
      *(float4*)(pred + (size_t)(0 * 2 + dir) * PS + tbase) =
          make_float4(chv[0], chv[1], chv[2], chv[3]);
      *(float4*)(pred + (size_t)(1 * 2 + dir) * PS + tbase) =
          make_float4(zhv[0], zhv[1], zhv[2], zhv[3]);
      *(float4*)(pred + (size_t)(2 * 2 + dir) * PS + tbase) =
          make_float4(xhv[0], xhv[1], xhv[2], xhv[3]);
#pragma unroll
      for (int j = 0; j < 4; ++j)
        atomicAdd(out + tbase + j, 0.5f * ccv[j]);  // mean over directions
    }
    __syncthreads();

    // stage E: gates = [c_c,m]@Wih.T + bih+bhh + hdec@Whh.T  (N=256, 8 waves x 2 tiles)
#pragma unroll
    for (int rep = 0; rep < 2; ++rep) {
      const int n0 = (wv + rep * 8) * 16;
      v8f c = tile_bias(sf + F_BIHH, n0, lane);
      c = wmma_gemm<128, 128>(c, sh + OFF_CAT2, sh + OFF_WIH, n0, lane);
      c = wmma_gemm<64, 64>(c, sh + OFF_HDEC, sh + OFF_WHH, n0, lane);
      tile_store<0>(sf + F_GATES, 256, n0, c, lane);
    }
    __syncthreads();

    // stage F: LSTM cell update (i,f,g,o splits of the 256-wide gates)
#pragma unroll
    for (int j = 0; j < 4; ++j) {
      const int hh = d0 + j;
      const float gi = sf[F_GATES + bl * 256 + hh];
      const float gf = sf[F_GATES + bl * 256 + 64 + hh];
      const float gg = sf[F_GATES + bl * 256 + 128 + hh];
      const float go = sf[F_GATES + bl * 256 + 192 + hh];
      const float cn = sigm(gf) * c4[j] + sigm(gi) * tanhf(gg);
      const float hn = sigm(go) * tanhf(cn);
      c4[j] = cn;
      sf[F_H32 + ei + j] = hn;
      sh[OFF_H16 + ei + j] = (h16)hn;
    }
    __syncthreads();
  }
}

extern "C" void kernel_launch(void* const* d_in, const int* in_sizes, int n_in,
                              void* d_out, int out_size, void* d_ws, size_t ws_size,
                              hipStream_t stream) {
  (void)in_sizes; (void)n_in; (void)out_size; (void)d_ws; (void)ws_size;
  const float* xg  = (const float*)d_in[0];
  const float* mg  = (const float*)d_in[1];
  const float* Wih = (const float*)d_in[2];
  const float* Whh = (const float*)d_in[3];
  const float* bih = (const float*)d_in[4];
  const float* bhh = (const float*)d_in[5];
  const float* Wdh = (const float*)d_in[6];
  const float* bdh = (const float*)d_in[7];
  const float* wdx = (const float*)d_in[8];
  const float* bdx = (const float*)d_in[9];
  const float* Whr = (const float*)d_in[10];
  const float* bhr = (const float*)d_in[11];
  const float* Wfr = (const float*)d_in[12];
  const float* bfr = (const float*)d_in[13];
  const float* Wwc = (const float*)d_in[14];
  const float* bwc = (const float*)d_in[15];

  float* out  = (float*)d_out;                       // [B,T,D,1]
  float* pred = out + (size_t)brits::B * brits::T * brits::D;  // [6,B,T,D,1]

  // out is accumulated atomically from both directions -> zero it each call.
  (void)hipMemsetAsync(d_out, 0,
      (size_t)brits::B * brits::T * brits::D * sizeof(float), stream);

  (void)hipFuncSetAttribute((const void*)brits_fused,
      hipFuncAttributeMaxDynamicSharedMemorySize, (int)brits::SMEM_BYTES);

  brits_fused<<<dim3(32), dim3(brits::NT), brits::SMEM_BYTES, stream>>>(
      xg, mg, Wih, Whh, bih, bhh, Wdh, bdh, wdx, bdx,
      Whr, bhr, Wfr, bfr, Wwc, bwc, out, pred);
}
